// CompressAttnMLA_88235808129222
// MI455X (gfx1250) — matrix operand order repaired and checked by reading, not compiled, not run on hardware
//
#include <hip/hip_runtime.h>
#include <hip/hip_bf16.h>
#include <stdint.h>

#define B_    2
#define S_    4096
#define H_    16
#define DQK_  192
#define NOPE_ 128
#define ROPE_ 64
#define DV_   128
#define LORA_ 512
#define KER_  32
#define STR_  16
#define SEL_  64
#define TOPK_ 16
#define NB_   255
#define NBP_  256
#define NSEL_ 64
#define HDV_  (H_ * DV_)           /* 2048 */
#define WKN_  (H_ * (NOPE_ + DV_)) /* 4096 */

typedef __attribute__((ext_vector_type(16))) __bf16        v16bf;
typedef __attribute__((ext_vector_type(8)))  float         v8f;
typedef __attribute__((ext_vector_type(8)))  unsigned int  v8u;

__device__ __forceinline__ unsigned short f2bf(float f) {
  unsigned int u = __float_as_uint(f);
  u += 0x7fffu + ((u >> 16) & 1u);           // round-to-nearest-even
  return (unsigned short)(u >> 16);
}
__device__ __forceinline__ unsigned int pk2(float lo, float hi) {
  return (unsigned int)f2bf(lo) | ((unsigned int)f2bf(hi) << 16);
}

// ---------------------------------------------------------------- k_zero
__global__ void k_zero(unsigned int* a, size_t na, unsigned int* b2, size_t nb) {
  size_t i = (size_t)blockIdx.x * blockDim.x + threadIdx.x;
  size_t stride = (size_t)gridDim.x * blockDim.x;
  for (size_t k = i; k < na + nb; k += stride) {
    if (k < na) a[k] = 0u; else b2[k - na] = 0u;
  }
}

// ------------------------------------------------------------- k_compress
// c_kv[b,n,d] = sum_k kv_lora[b, n*16+k, d] * w_cmp_kv[k]   (and c_pe likewise)
__global__ void k_compress(const float* __restrict__ kvl, const float* __restrict__ kpe,
                           const float* __restrict__ wck, const float* __restrict__ wcp,
                           float* __restrict__ ckv, float* __restrict__ cpe) {
  const int NKV = B_ * NB_ * LORA_;
  const int NPE = B_ * NB_ * ROPE_;
  int tid = blockIdx.x * blockDim.x + threadIdx.x;
  if (tid < NKV) {
    int b = tid / (NB_ * LORA_);
    int rem = tid - b * (NB_ * LORA_);
    int n = rem / LORA_, d = rem - n * LORA_;
    const float* src = kvl + ((size_t)b * S_ + n * STR_) * LORA_ + d;
    float s = 0.f;
#pragma unroll 1
    for (int k = 0; k < KER_; ++k) s += src[(size_t)k * LORA_] * wck[k];
    ckv[tid] = s;
  } else if (tid < NKV + NPE) {
    int t2 = tid - NKV;
    int b = t2 / (NB_ * ROPE_);
    int rem = t2 - b * (NB_ * ROPE_);
    int n = rem / ROPE_, d = rem - n * ROPE_;
    const float* src = kpe + ((size_t)b * S_ + n * STR_) * ROPE_ + d;
    float s = 0.f;
#pragma unroll 1
    for (int k = 0; k < KER_; ++k) s += src[(size_t)k * ROPE_] * wcp[k];
    cpe[t2] = s;
  }
}

// -------------------------------------------------------------- k_prep_q
// q (B,S,H,192) f32  ->  qb [b][h][s][192] bf16  (16-row tiles contiguous)
__global__ void k_prep_q(const float* __restrict__ q, unsigned short* __restrict__ qb) {
  size_t tid = (size_t)blockIdx.x * blockDim.x + threadIdx.x;
  const size_t N = (size_t)B_ * H_ * S_ * DQK_;
  if (tid >= N) return;
  int d = (int)(tid % DQK_);
  size_t r = tid / DQK_;
  int s = (int)(r % S_); r /= S_;
  int h = (int)(r % H_); int b = (int)(r / H_);
  qb[tid] = f2bf(q[(((size_t)b * S_ + s) * H_ + h) * DQK_ + d]);
}

// ---------------------------------------------------------------- k_rope
// cmp_k[b][n][h][128+j] = c_pe[b][n][j]  (broadcast over heads)
__global__ void k_rope(const float* __restrict__ cpe, unsigned short* __restrict__ ck) {
  int tid = blockIdx.x * blockDim.x + threadIdx.x;
  const int N = B_ * NB_ * H_ * ROPE_;
  if (tid >= N) return;
  int j = tid & (ROPE_ - 1);
  int r = tid >> 6;
  int h = r & (H_ - 1); r >>= 4;
  int n = r % NB_; int b = r / NB_;
  ck[(((size_t)(b * NBP_ + n)) * H_ + h) * DQK_ + NOPE_ + j] =
      f2bf(cpe[((size_t)b * NB_ + n) * ROPE_ + j]);
}

// ----------------------------------------------------------------- k_kvb
// kv = c_kv(255x512) @ w_kv_b(512x4096); scatter into cmp_k (nope) and v, bf16.
// One wave per 16x16 output tile; K=512 over 16 bf16 WMMA steps.
// A-row index is CLAMPED (not guarded): garbage rows feed only D-rows >= 255
// which are never stored, so gathers stay unconditional (no exec churn).
__global__ __launch_bounds__(32) void k_kvb(const float* __restrict__ ckv,
                                            const float* __restrict__ wkb,
                                            unsigned short* __restrict__ ck,
                                            unsigned short* __restrict__ vv) {
  const int nt = blockIdx.x, mt = blockIdx.y, b = blockIdx.z;
  const int lane = threadIdx.x, col = lane & 15, half = lane >> 4;
  const int m = mt * 16 + col;                       // logical A-row
  const int mrow = (m < NB_) ? m : (NB_ - 1);        // clamped gather row
  const int ncol = nt * 16 + col;                    // B/D column this lane owns
  const float* ap = ckv + ((size_t)(b * NB_ + mrow)) * LORA_;
  const float* bp = wkb + ncol;
  v8f c = {};
#pragma unroll 1
  for (int kk = 0; kk < 16; ++kk) {
    v8u au, bu;
#pragma unroll
    for (int j = 0; j < 8; ++j) {
      // A fragment (16x32 bf16): K = half*16 + 2j (+1)
      int ka = kk * 32 + half * 16 + 2 * j;
      au[j] = pk2(ap[ka], ap[ka + 1]);
      // B fragment (32x16 bf16): K = half*16 + 2j (+1), consecutive rows of wkb
      int kb = kk * 32 + half * 16 + 2 * j;
      bu[j] = pk2(bp[(size_t)kb * WKN_], bp[(size_t)(kb + 1) * WKN_]);
    }
    c = __builtin_amdgcn_wmma_f32_16x16x32_bf16(
        false, __builtin_bit_cast(v16bf, au),
        false, __builtin_bit_cast(v16bf, bu),
        (short)0, c, false, false);
  }
  const int hh = ncol >> 8, cc = ncol & 255;
#pragma unroll
  for (int r = 0; r < 8; ++r) {
    int mm = mt * 16 + r + 8 * half;   // D row = r + 8*half
    if (mm < NB_) {
      unsigned short val = f2bf(c[r]);
      if (cc < NOPE_)
        ck[(((size_t)(b * NBP_ + mm)) * H_ + hh) * DQK_ + cc] = val;
      else
        vv[(((size_t)(b * NBP_ + mm)) * H_ + hh) * DV_ + (cc - NOPE_)] = val;
    }
  }
}

// ----------------------------------------------------------------- k_attn
// One wave per (b, h, 16-row s-tile). Full 16x256 score tile held in 16 v8f
// accumulators; masked softmax in registers; pg via float-bits atomicMax;
// p->bf16 in LDS; o = p@v with v staged transposed in LDS.
__global__ __launch_bounds__(32) void k_attn(const unsigned short* __restrict__ qb,
                                             const unsigned short* __restrict__ ckg,
                                             const unsigned short* __restrict__ vv,
                                             int* __restrict__ pg,
                                             float* __restrict__ out) {
  const int st = blockIdx.x, h = blockIdx.y, b = blockIdx.z;
  const int lane = threadIdx.x;
  const int col = lane & 15, half = lane >> 4;
  const int s0 = st * 16;
  const float scl = 0.07216878364870323f;   // 192^-0.5

  __shared__ __align__(16) unsigned short qs[16 * 192];
  __shared__ __align__(16) unsigned short ks[16 * 192];
  __shared__ __align__(16) unsigned short psm[16 * 256];
  __shared__ __align__(16) unsigned short vts[128 * 32];   // [dv][blk] transposed

  // stage q tile: 1536 contiguous dwords -> 384 b128 transfers (12/lane)
  {
    const uint4* qg4 = (const uint4*)(qb + (((size_t)(b * H_ + h)) * S_ + s0) * DQK_);
    uint4* qd4 = (uint4*)qs;
#pragma unroll 1
    for (int i = 0; i < 12; ++i) qd4[i * 32 + lane] = qg4[i * 32 + lane];
  }
  // zero p buffer with b128 stores (covers tiles beyond ntiles)
  {
    uint4 z; z.x = z.y = z.z = z.w = 0u;
    uint4* pd4 = (uint4*)psm;
#pragma unroll 1
    for (int i = 0; i < 16; ++i) pd4[i * 32 + lane] = z;
  }

  int ntiles = 0;
  { int tmax = s0 + 15; if (tmax >= KER_ - 1) ntiles = (((tmax - (KER_ - 1)) >> 4) >> 4) + 1; }

  v8f accv[16] = {};

  // -------- score phase: scores[s,n] = q . cmp_k, K=192 in 6 bf16 steps
#pragma unroll
  for (int nt = 0; nt < 16; ++nt) {
    if (nt < ntiles) {
      if (nt + 1 < ntiles)
        __builtin_prefetch(ckg + (((size_t)(b * NBP_ + (nt + 1) * 16)) * H_ + h) * DQK_, 0, 1);
      // stage cmp_k tile: 16 rows x 24 dqwords (lanes 0..23 active)
      const uint4* kg4 =
          (const uint4*)(ckg + (((size_t)(b * NBP_ + nt * 16)) * H_ + h) * DQK_);
      uint4* kd4 = (uint4*)ks;
      if (lane < 24) {
#pragma unroll 1
        for (int blk = 0; blk < 16; ++blk)
          kd4[blk * 24 + lane] = kg4[(size_t)blk * (H_ * 24) + lane];
      }
      asm volatile("s_wait_dscnt 0" ::: "memory");
      v8f c = accv[nt];
#pragma unroll
      for (int kk = 0; kk < 6; ++kk) {
        v8u au, bu;
#pragma unroll
        for (int j = 0; j < 8; ++j) {
          int ka = kk * 32 + half * 16 + 2 * j;
          au[j] = *(const unsigned int*)&qs[col * DQK_ + ka];       // A: M=col
          int kb = kk * 32 + half * 16 + 2 * j;
          bu[j] = *(const unsigned int*)&ks[col * DQK_ + kb];       // B: N=col(block)
        }
        c = __builtin_amdgcn_wmma_f32_16x16x32_bf16(
            false, __builtin_bit_cast(v16bf, au),
            false, __builtin_bit_cast(v16bf, bu),
            (short)0, c, false, false);
      }
      accv[nt] = c;
    }
  }

  // -------- masked softmax, rows distributed as r + 8*half across halves
  float inv[8];
#pragma unroll
  for (int r = 0; r < 8; ++r) {
    const int t = s0 + r + 8 * half;
    float pm = -3.0e38f;
#pragma unroll
    for (int nt = 0; nt < 16; ++nt) {
      if (nt < ntiles) {
        int n = nt * 16 + col;
        float sv = (t >= n * 16 + (KER_ - 1)) ? accv[nt][r] * scl : -3.0e38f;
        accv[nt][r] = sv;
        pm = fmaxf(pm, sv);
      }
    }
#pragma unroll
    for (int mk = 1; mk < 16; mk <<= 1) pm = fmaxf(pm, __shfl_xor(pm, mk, 32));
    float ls = 0.0f;
#pragma unroll
    for (int nt = 0; nt < 16; ++nt) {
      if (nt < ntiles) {
        float sv = accv[nt][r];
        float e = (sv > -1.0e37f) ? __expf(sv - pm) : 0.0f;
        accv[nt][r] = e;
        ls += e;
      }
    }
#pragma unroll
    for (int mk = 1; mk < 16; mk <<= 1) ls += __shfl_xor(ls, mk, 32);
    inv[r] = (ls > 0.0f) ? (1.0f / ls) : 0.0f;
  }

  // -------- p -> LDS (bf16) and pg = max over heads via float-bits atomicMax
  int* pgb = pg + (size_t)b * S_ * NBP_;
#pragma unroll
  for (int nt = 0; nt < 16; ++nt) {
    if (nt < ntiles) {
#pragma unroll
      for (int r = 0; r < 8; ++r) {
        const int t = s0 + r + 8 * half;
        float p = accv[nt][r] * inv[r];
        psm[(r + 8 * half) * 256 + nt * 16 + col] = f2bf(p);
        if (p > 0.0f)
          atomicMax(&pgb[(size_t)t * NBP_ + nt * 16 + col], __float_as_int(p));
      }
    }
  }
  asm volatile("s_wait_dscnt 0" ::: "memory");

  // -------- o = p @ v : K over blocks in chunks of 32, N over dv in 8 chunks
  v8f oa[8] = {};
#pragma unroll
  for (int c2 = 0; c2 < 8; ++c2) {
    if (c2 * 2 < ntiles) {
      // stage v chunk transposed: lane owns dv = lane*4 .. lane*4+3
      const unsigned short* vgb = vv + (((size_t)(b * NBP_ + c2 * 32)) * H_ + h) * DV_;
      const int dvb = lane * 4;
#pragma unroll 1
      for (int blk = 0; blk < 32; ++blk) {
        uint2 w2 = *(const uint2*)(vgb + (size_t)blk * (H_ * DV_) + dvb);
        vts[(dvb + 0) * 32 + blk] = (unsigned short)(w2.x & 0xffffu);
        vts[(dvb + 1) * 32 + blk] = (unsigned short)(w2.x >> 16);
        vts[(dvb + 2) * 32 + blk] = (unsigned short)(w2.y & 0xffffu);
        vts[(dvb + 3) * 32 + blk] = (unsigned short)(w2.y >> 16);
      }
      asm volatile("s_wait_dscnt 0" ::: "memory");
#pragma unroll
      for (int dc = 0; dc < 8; ++dc) {
        v8u au, bu;
#pragma unroll
        for (int j = 0; j < 8; ++j) {
          int ka = half * 16 + 2 * j;      // K = block
          au[j] = *(const unsigned int*)&psm[col * 256 + c2 * 32 + ka];
          int kb = half * 16 + 2 * j;
          bu[j] = *(const unsigned int*)&vts[(dc * 16 + col) * 32 + kb];
        }
        oa[dc] = __builtin_amdgcn_wmma_f32_16x16x32_bf16(
            false, __builtin_bit_cast(v16bf, au),
            false, __builtin_bit_cast(v16bf, bu),
            (short)0, oa[dc], false, false);
      }
    }
  }

  float* ob = out + (size_t)b * S_ * HDV_ + h * DV_;
#pragma unroll
  for (int dc = 0; dc < 8; ++dc)
#pragma unroll
    for (int r = 0; r < 8; ++r) {
      int s = s0 + r + 8 * half;
      ob[(size_t)s * HDV_ + dc * 16 + col] = oa[dc][r];
    }
}

// ----------------------------------------------------------------- k_topk
// sel[m] = 0.5*pg[4m-1] + pg[4m]+pg[4m+1]+pg[4m+2] + 0.5*pg[4m+3] (banded M)
__global__ void k_topk(const float* __restrict__ pg, int* __restrict__ ti) {
  int tid = blockIdx.x * blockDim.x + threadIdx.x;
  if (tid >= B_ * S_) return;
  int s = tid & (S_ - 1);
  const float* row = pg + (size_t)tid * NBP_;
  const float BIGV = 1.0e30f, NEGBIG = -1.0e30f;
  float sel[NSEL_];
  int c = s >> 6;
#pragma unroll 1
  for (int m2 = 0; m2 < NSEL_; ++m2) {
    float a = row[4 * m2] + row[4 * m2 + 1] + row[4 * m2 + 2];
    if (4 * m2 - 1 >= 0) a += 0.5f * row[4 * m2 - 1];
    if (4 * m2 + 3 <= NB_ - 1) a += 0.5f * row[4 * m2 + 3];
    bool causal = (m2 <= c);
    bool forced = causal && ((m2 < 1) || (m2 >= c - 1));
    sel[m2] = causal ? (forced ? BIGV : a) : NEGBIG;
  }
  int* o = ti + (size_t)tid * TOPK_;
#pragma unroll 1
  for (int t = 0; t < TOPK_; ++t) {
    float best = -3.5e38f; int bi = 0;
#pragma unroll 1
    for (int m2 = 0; m2 < NSEL_; ++m2)
      if (sel[m2] > best) { best = sel[m2]; bi = m2; }
    o[t] = (best > -5.0e29f) ? bi : -1;
    sel[bi] = -3.5e38f;
  }
}

// -------------------------------------------------------------------------
extern "C" void kernel_launch(void* const* d_in, const int* in_sizes, int n_in,
                              void* d_out, int out_size, void* d_ws, size_t ws_size,
                              hipStream_t stream) {
  (void)in_sizes; (void)n_in; (void)out_size; (void)ws_size;
  const float* q   = (const float*)d_in[0];
  const float* kvl = (const float*)d_in[1];
  const float* kpe = (const float*)d_in[2];
  const float* wkb = (const float*)d_in[3];
  const float* wck = (const float*)d_in[4];
  const float* wcp = (const float*)d_in[5];

  char* ws = (char*)d_ws;
  const size_t OFF_CKV = 0;                       // 2*255*512*4   = 1,044,480
  const size_t OFF_CPE = 1048576;                 // 2*255*64*4    =   130,560
  const size_t OFF_CK  = OFF_CPE + 131072;        // bf16 2*256*16*192 = 3,145,728
  const size_t OFF_V   = OFF_CK + 3145728;        // bf16 2*256*16*128 = 2,097,152
  const size_t OFF_QB  = OFF_V + 2097152;         // bf16 2*16*4096*192 = 50,331,648
  const size_t OFF_PG  = OFF_QB + 50331648;       // f32 2*4096*256 = 8,388,608
  float*          ckv = (float*)(ws + OFF_CKV);
  float*          cpe = (float*)(ws + OFF_CPE);
  unsigned short* ck  = (unsigned short*)(ws + OFF_CK);
  unsigned short* vv  = (unsigned short*)(ws + OFF_V);
  unsigned short* qb  = (unsigned short*)(ws + OFF_QB);
  int*            pg  = (int*)(ws + OFF_PG);

  // zero cmp_k+v (contiguous) and pg
  k_zero<<<2048, 256, 0, stream>>>((unsigned int*)ck, (3145728u + 2097152u) / 4,
                                   (unsigned int*)pg, 8388608u / 4);

  int ncmp = B_ * NB_ * (LORA_ + ROPE_);
  k_compress<<<(ncmp + 255) / 256, 256, 0, stream>>>(kvl, kpe, wck, wcp, ckv, cpe);

  size_t nq = (size_t)B_ * H_ * S_ * DQK_;
  k_prep_q<<<(unsigned)((nq + 255) / 256), 256, 0, stream>>>(q, qb);

  int nrope = B_ * NB_ * H_ * ROPE_;
  k_rope<<<(nrope + 255) / 256, 256, 0, stream>>>(cpe, ck);

  k_kvb<<<dim3(WKN_ / 16, 16, B_), 32, 0, stream>>>(ckv, wkb, ck, vv);

  k_attn<<<dim3(S_ / 16, H_, B_), 32, 0, stream>>>(qb, ck, vv, pg, (float*)d_out);

  int* ti = (int*)d_out + (size_t)B_ * S_ * HDV_;
  k_topk<<<(B_ * S_ + 255) / 256, 256, 0, stream>>>((const float*)pg, ti);
}
